// DivFreeNetwork_14379550507511
// MI455X (gfx1250) — matrix operand order repaired and checked by reading, not compiled, not run on hardware
//
#include <hip/hip_runtime.h>
#include <hip/hip_bf16.h>
#include <math.h>

typedef __attribute__((ext_vector_type(16))) _Float16 v16h;
typedef __attribute__((ext_vector_type(8)))  float    v8f;

// gcc-style int4 vector to match the async-load builtin's pointee type
typedef int i32x4 __attribute__((__vector_size__(16)));
typedef __attribute__((address_space(1))) i32x4* as1_i32x4p;
typedef __attribute__((address_space(3))) i32x4* as3_i32x4p;

#define DM 1024          // D_MODEL
#define NMIX 64
#define TRI 6
#define ODIM 448         // NMIX*(TRI+1)

// CDNA5 async global->LDS copy path (ASYNCcnt-tracked), with fallback to
// register-staged copies if the toolchain does not declare the builtins.
#if defined(__HIP_DEVICE_COMPILE__) && \
    __has_builtin(__builtin_amdgcn_global_load_async_to_lds_b128) && \
    __has_builtin(__builtin_amdgcn_s_wait_asynccnt)
#define USE_ASYNC_LDS 1
#else
#define USE_ASYNC_LDS 0
#endif

// ---------------------------------------------------------------------------
// Weight conversion fp32 -> f16 WITH TRANSPOSE:
//   Whf[l][n][k] = Wh[l][k][n]   (4 x 1024 x 1024)
//   Wof[n][k]    = Wout[k][n]    (448 x 1024)
// so GEMM B-tiles can be staged with pure b128 loads/stores (K contiguous).
// ---------------------------------------------------------------------------
__global__ void k_convert(const float* __restrict__ Wh, const float* __restrict__ Wout,
                          _Float16* __restrict__ Whf, _Float16* __restrict__ Wof) {
    int i = blockIdx.x * blockDim.x + threadIdx.x;
    const int nWh = 4 * DM * DM;
    if (i < nWh) {
        int l = i >> 20;              // / (DM*DM)
        int r = i & (DM * DM - 1);
        int k = r >> 10;
        int n = r & (DM - 1);
        Whf[(size_t)l * DM * DM + (size_t)n * DM + k] = (_Float16)Wh[i];
    } else {
        int j = i - nWh;
        if (j < DM * ODIM) {
            int k = j / ODIM;
            int n = j - k * ODIM;
            Wof[(size_t)n * DM + k] = (_Float16)Wout[j];
        }
    }
}

// ---------------------------------------------------------------------------
// Input layer: pre = x@W0 + b0 (K=4).  Primal h0 = silu(pre),
// tangent seeds t_j = silu'(pre) * W0[j,:]  (since e_j @ W0 = W0 row j).
// H layout: channel-major [5][nS][1024], f16.
// ---------------------------------------------------------------------------
__global__ void k_input(const float* __restrict__ x, const float* __restrict__ W0,
                        const float* __restrict__ b0, _Float16* __restrict__ H, int nS) {
    int i = blockIdx.x * blockDim.x + threadIdx.x;
    if (i >= nS * DM) return;
    int s = i >> 10;
    int n = i & (DM - 1);
    const float* xs = x + (size_t)s * 4;
    float w0n[4];
    float pre = b0[n];
#pragma unroll
    for (int k = 0; k < 4; ++k) {
        w0n[k] = W0[k * DM + n];
        pre += xs[k] * w0n[k];
    }
    float sig = 1.0f / (1.0f + expf(-pre));
    float h0  = pre * sig;
    float sp  = sig * (1.0f + pre * (1.0f - sig));   // silu'(pre)
    size_t cs = (size_t)nS * DM;
    H[(size_t)s * DM + n] = (_Float16)h0;
#pragma unroll
    for (int j = 0; j < 4; ++j)
        H[cs * (j + 1) + (size_t)s * DM + n] = (_Float16)(sp * w0n[j]);
}

// ---------------------------------------------------------------------------
// WMMA GEMM for one layer, all 5 channels per wave, NFRAG B-fragments/wave.
// Hin: [5][nS][1024] f16 channel-major.  Wt: [N][1024] f16 (pre-transposed).
// Block = 256 threads (8 waves).  Block tile: 32 samples x (64*NFRAG) cols.
// Wave tile: 16 samples x (16*NFRAG) cols x 5 channels.
// Double-buffered LDS; next K-tile copied global->LDS asynchronously
// (ASYNCcnt) while the current tile feeds the WMMA pipe.
// HIDDEN epilogue: h0 = silu(z0), t_c = silu'(z0) * acc_c   (f16 out)
// OUT    epilogue: primal z0 = acc0 + bout, tangents raw    (f32 out)
// ---------------------------------------------------------------------------
template <bool HIDDEN, int NFRAG>
__global__ void k_gemm(const _Float16* __restrict__ Hin, const _Float16* __restrict__ Wt,
                       const float* __restrict__ bias, _Float16* __restrict__ HoutH,
                       float* __restrict__ HoutF, int nS) {
    constexpr int NT    = 64 * NFRAG;      // cols per block
    constexpr int NB4_A = 5 * 32 * 32 / 8; // 640 uint4 per A tile
    constexpr int NB4_B = NT * 4;          // uint4 per B tile

    __shared__ __align__(16) _Float16 As[2][5][32][32];  // [buf][chan][sample][k]
    __shared__ __align__(16) _Float16 Bs[2][NT][32];     // [buf][col][k]

    const int tid   = threadIdx.x;
    const int lane  = tid & 31;
    const int wave  = tid >> 5;
    const int wm    = wave >> 2;            // 0..1 sample sub-block
    const int wn    = wave & 3;             // 0..3 col group
    const int nBase = blockIdx.x * NT;
    const int sBase = blockIdx.y * 32;
    const size_t chIn = (size_t)nS * DM;

    v8f acc[5][NFRAG] = {};

#if USE_ASYNC_LDS
    // ---- async global->LDS copy of one 32-wide K tile into buffer `buf` ----
    auto fetchTile = [&](int k0, int buf) {
#pragma unroll
        for (int t = 0; t < 3; ++t) {
            int i = tid + t * 256;
            if (i < NB4_A) {
                int c = i >> 7, rem = i & 127, m = rem >> 2, kq = (rem & 3) << 3;
                __builtin_amdgcn_global_load_async_to_lds_b128(
                    (as1_i32x4p)(Hin + c * chIn + (size_t)(sBase + m) * DM + k0 + kq),
                    (as3_i32x4p)(&As[buf][c][m][kq]),
                    0, 0);
            }
        }
#pragma unroll
        for (int t = 0; t < 2; ++t) {
            int i = tid + t * 256;
            if (i < NB4_B) {
                int n = i >> 2, kq = (i & 3) << 3;
                __builtin_amdgcn_global_load_async_to_lds_b128(
                    (as1_i32x4p)(Wt + (size_t)(nBase + n) * DM + k0 + kq),
                    (as3_i32x4p)(&Bs[buf][n][kq]),
                    0, 0);
            }
        }
    };

    fetchTile(0, 0);
    __builtin_amdgcn_s_wait_asynccnt(0);
    __syncthreads();
#else
    uint4 ra[3], rb[2];
    auto fetchA = [&](int k0) {
#pragma unroll
        for (int t = 0; t < 3; ++t) {
            int i = tid + t * 256;
            if (i < NB4_A) {
                int c = i >> 7, rem = i & 127, m = rem >> 2, kq = (rem & 3) << 3;
                ra[t] = *reinterpret_cast<const uint4*>(
                    Hin + c * chIn + (size_t)(sBase + m) * DM + k0 + kq);
            }
        }
#pragma unroll
        for (int t = 0; t < 2; ++t) {
            int i = tid + t * 256;
            if (i < NB4_B) {
                int n = i >> 2, kq = (i & 3) << 3;
                rb[t] = *reinterpret_cast<const uint4*>(
                    Wt + (size_t)(nBase + n) * DM + k0 + kq);
            }
        }
    };
    auto stash = [&](int buf) {
#pragma unroll
        for (int t = 0; t < 3; ++t) {
            int i = tid + t * 256;
            if (i < NB4_A) {
                int c = i >> 7, rem = i & 127, m = rem >> 2, kq = (rem & 3) << 3;
                *reinterpret_cast<uint4*>(&As[buf][c][m][kq]) = ra[t];
            }
        }
#pragma unroll
        for (int t = 0; t < 2; ++t) {
            int i = tid + t * 256;
            if (i < NB4_B) {
                int n = i >> 2, kq = (i & 3) << 3;
                *reinterpret_cast<uint4*>(&Bs[buf][n][kq]) = rb[t];
            }
        }
    };

    fetchA(0);
    stash(0);
    __syncthreads();
#endif

    const int m16 = lane & 15;
    const int kg  = (lane >> 4) << 3;   // A half-wave K offset: 0 or 8
    const int kb  = (lane >> 4) << 4;   // unused (B fragments also use kg): 0 or 16

    int p = 0;
    for (int step = 0; step < DM / 32; ++step) {
        const bool more = (step + 1 < DM / 32);
#if USE_ASYNC_LDS
        if (more) fetchTile((step + 1) * 32, p ^ 1);  // in flight under WMMA
#else
        if (more) fetchA((step + 1) * 32);
#endif

        // B fragments: lane = col (mod 16), K kg..kg+7 and 16+kg..16+kg+7
        union { uint4 u[2]; v16h h; } bf[NFRAG];
#pragma unroll
        for (int f = 0; f < NFRAG; ++f) {
            const uint4* bp = reinterpret_cast<const uint4*>(
                &Bs[p][wn * (16 * NFRAG) + f * 16 + m16][kg]);
            bf[f].u[0] = bp[0];
            bf[f].u[1] = bp[2];
        }
#pragma unroll
        for (int c = 0; c < 5; ++c) {
            union { uint4 u[2]; v16h h; } af;
            const _Float16* arow = &As[p][c][wm * 16 + m16][0];
            af.u[0] = *reinterpret_cast<const uint4*>(arow + kg);       // K kg..kg+7
            af.u[1] = *reinterpret_cast<const uint4*>(arow + 16 + kg);  // K 16+kg..
#pragma unroll
            for (int f = 0; f < NFRAG; ++f)
                acc[c][f] = __builtin_amdgcn_wmma_f32_16x16x32_f16(
                    false, af.h, false, bf[f].h, (short)0, acc[c][f], false, false);
        }

#if USE_ASYNC_LDS
        if (more) __builtin_amdgcn_s_wait_asynccnt(0);
#else
        if (more) stash(p ^ 1);
#endif
        __syncthreads();
        p ^= 1;
    }

    // ---- epilogue: (VGPR r, lane) -> M = r + 8*(lane>=16), Ncol = lane&15 ----
    const size_t chOutH = (size_t)nS * DM;
    const size_t chOutF = (size_t)nS * ODIM;
#pragma unroll
    for (int f = 0; f < NFRAG; ++f) {
        const int nG = nBase + wn * (16 * NFRAG) + f * 16 + (lane & 15);
        const float bn = bias[nG];
#pragma unroll
        for (int r = 0; r < 8; ++r) {
            size_t s = (size_t)(sBase + wm * 16 + r + ((lane >> 4) << 3));
            float z0 = acc[0][f][r] + bn;
            if (HIDDEN) {
                float sig = 1.0f / (1.0f + expf(-z0));
                float h0  = z0 * sig;
                float sp  = sig * (1.0f + z0 * (1.0f - sig));
                HoutH[s * DM + nG] = (_Float16)h0;
#pragma unroll
                for (int c = 1; c < 5; ++c)
                    HoutH[c * chOutH + s * DM + nG] = (_Float16)(sp * acc[c][f][r]);
            } else {
                HoutF[s * ODIM + nG] = z0;                          // primal gets bout
#pragma unroll
                for (int c = 1; c < 5; ++c)
                    HoutF[c * chOutF + s * ODIM + nG] = acc[c][f][r]; // tangents: no bias
            }
        }
    }
}

// ---------------------------------------------------------------------------
// Final: softmax mixture + JVP + antisymmetric divergence.
// OUT: [5][nS][448] f32.  out: [nS][4]  = [rho*10, flux_xyz].
// ---------------------------------------------------------------------------
__global__ void k_final(const float* __restrict__ OUT, float* __restrict__ out, int nS) {
    int s = blockIdx.x * blockDim.x + threadIdx.x;
    if (s >= nS) return;
    const size_t cs = (size_t)nS * ODIM;
    const float* o0 = OUT + (size_t)s * ODIM;

    float mx = -INFINITY;
    for (int m = 0; m < NMIX; ++m) mx = fmaxf(mx, o0[m]);
    float se = 0.0f;
    for (int m = 0; m < NMIX; ++m) se += expf(o0[m] - mx);
    float inv = 1.0f / se;

    float dot[4] = {0.f, 0.f, 0.f, 0.f};
    for (int m = 0; m < NMIX; ++m) {
        float p = expf(o0[m] - mx) * inv;
#pragma unroll
        for (int j = 0; j < 4; ++j)
            dot[j] += p * OUT[cs * (j + 1) + (size_t)s * ODIM + m];
    }

    float da[4][TRI] = {};
    for (int m = 0; m < NMIX; ++m) {
        float p = expf(o0[m] - mx) * inv;
        float vals[TRI];
#pragma unroll
        for (int k = 0; k < TRI; ++k) vals[k] = o0[NMIX + m * TRI + k];
#pragma unroll
        for (int j = 0; j < 4; ++j) {
            const float* oj = OUT + cs * (j + 1) + (size_t)s * ODIM;
            float dl   = oj[m];
            float coef = p * (dl - dot[j]);      // dp_m = p_m (dl_m - p.dl)
#pragma unroll
            for (int k = 0; k < TRI; ++k)
                da[j][k] += coef * vals[k] + p * oj[NMIX + m * TRI + k];
        }
    }
    // u_i = sum_j dA_ij/dz_j with A upper-tri pairs (01,02,03,12,13,23)
    float u0 =  da[1][0] + da[2][1] + da[3][2];
    float u1 = -da[0][0] + da[2][3] + da[3][4];
    float u2 = -da[0][1] - da[1][3] + da[3][5];
    float u3 = -da[0][2] - da[1][4] - da[2][5];
    float4 r = {u0 * 10.0f, u1, u2, u3};
    *reinterpret_cast<float4*>(out + (size_t)s * 4) = r;
}

// ---------------------------------------------------------------------------
extern "C" void kernel_launch(void* const* d_in, const int* in_sizes, int n_in,
                              void* d_out, int out_size, void* d_ws, size_t ws_size,
                              hipStream_t stream) {
    const float* x    = (const float*)d_in[0];
    const float* W0   = (const float*)d_in[1];
    const float* b0   = (const float*)d_in[2];
    const float* Wh   = (const float*)d_in[3];
    const float* bh   = (const float*)d_in[4];
    const float* Wout = (const float*)d_in[5];
    const float* bout = (const float*)d_in[6];
    float* out = (float*)d_out;
    const int B = in_sizes[0] / 4;

    // Workspace layout
    char* ws = (char*)d_ws;
    const size_t offWh = 0;
    const size_t offWo = (size_t)4 * DM * DM * 2;        // 8 MB  (Whf, transposed)
    const size_t offH  = offWo + (size_t)ODIM * DM * 2;  // + ~0.9 MB (Wof, transposed)
    const size_t perSample = 2 * (5 * DM * 2) + 5 * ODIM * 4;  // HA+HB f16, OUT f32

    int chunk = 65536;
    if (chunk > B) chunk = B;
    while (chunk > 256 && offH + (size_t)chunk * perSample > ws_size) chunk >>= 1;

    _Float16* Whf = (_Float16*)(ws + offWh);
    _Float16* Wof = (_Float16*)(ws + offWo);
    _Float16* HA  = (_Float16*)(ws + offH);
    _Float16* HB  = HA + (size_t)5 * chunk * DM;
    float*    OUT = (float*)(HB + (size_t)5 * chunk * DM);

    // Convert + transpose weights once per launch
    const int nconv = 4 * DM * DM + DM * ODIM;
    k_convert<<<(nconv + 255) / 256, 256, 0, stream>>>(Wh, Wout, Whf, Wof);

    for (int s0 = 0; s0 < B; s0 += chunk) {
        int cur = chunk;  // B and chunk are powers of two; chunk divides B
        k_input<<<(cur * DM + 255) / 256, 256, 0, stream>>>(x + (size_t)s0 * 4, W0, b0, HA, cur);

        for (int l = 0; l < 4; ++l) {
            const _Float16* hin = (l & 1) ? HB : HA;
            _Float16*       ho  = (l & 1) ? HA : HB;
            k_gemm<true, 2><<<dim3(DM / 128, cur / 32), 256, 0, stream>>>(
                hin, Whf + (size_t)l * DM * DM, bh + l * DM, ho, nullptr, cur);
        }
        // After 4 layers, last hidden activations are in HA
        k_gemm<false, 1><<<dim3(ODIM / 64, cur / 32), 256, 0, stream>>>(
            HA, Wof, bout, nullptr, OUT, cur);

        k_final<<<(cur + 255) / 256, 256, 0, stream>>>(OUT, out + (size_t)s0 * 4, cur);
    }
}
